// MultiHeadStructureAttention_19688130085318
// MI455X (gfx1250) — matrix operand order repaired and checked by reading, not compiled, not run on hardware
//
#include <hip/hip_runtime.h>
#include <hip/hip_bf16.h>
#include <math.h>

// ---------------------------------------------------------------------------
// Types for CDNA5 WMMA (wave32): D(16x16 f32) = A(16x32 bf16) * B(32x16 bf16) + C
// ---------------------------------------------------------------------------
typedef __bf16 bf16_t;
typedef __attribute__((ext_vector_type(16))) __bf16 v16bf;
typedef __attribute__((ext_vector_type(8)))  float  v8f;

#define BB 4
#define SS 2048
#define DD 1024
#define HH 16
#define DK 64
#define DV 64
#define MROWS (BB * SS)   // 8192

// Workspace layout (bytes)
#define WQT_OFF   0u
#define WKT_OFF   (2u * 1024u * 1024u)
#define WVT_OFF   (4u * 1024u * 1024u)
#define WFCT_OFF  (6u * 1024u * 1024u)
#define QH_OFF    (8u * 1024u * 1024u)                    // 16 MB bf16
#define KH_OFF    (QH_OFF + 16u * 1024u * 1024u)          // 16 MB bf16
#define VT_OFF    (KH_OFF + 16u * 1024u * 1024u)          // 16 MB bf16
#define ATTN_OFF  (VT_OFF + 16u * 1024u * 1024u)          // 32 MB f32
#define FC_OFF    QH_OFF  // alias: Qh/Kh dead once attention kernel finished

// ---------------------------------------------------------------------------
// WMMA helpers
// ---------------------------------------------------------------------------
__device__ __forceinline__ v8f wmma_bf16(v16bf a, v16bf b, v8f c) {
  // (neg_a, A, neg_b, B, c_mod, C, reuse_a, reuse_b)
  return __builtin_amdgcn_wmma_f32_16x16x32_bf16(false, a, false, b,
                                                 (short)0, c, false, false);
}

// A fragment (16x32, M x K) from row-major bf16 matrix.
// lane l: m = l&15, khalf = l>>4; elems 0..7 -> K = khalf*8 + j,
// elems 8..15 -> K = 16 + khalf*8 + j   (ISA 7.12.2, 16-bit A 16x32)
__device__ __forceinline__ v16bf load_a_bf16(const bf16_t* __restrict__ base,
                                             int ld, int m0, int k0, int lane) {
  int m = lane & 15, kh = lane >> 4;
  const bf16_t* p = base + (size_t)(m0 + m) * ld + k0;
  v16bf a;
#pragma unroll
  for (int j = 0; j < 8; ++j) a[j] = p[kh * 8 + j];
#pragma unroll
  for (int j = 0; j < 8; ++j) a[8 + j] = p[16 + kh * 8 + j];
  return a;
}

// A fragment from row-major fp32 matrix with on-the-fly scale + cvt to bf16.
__device__ __forceinline__ v16bf load_a_f32(const float* __restrict__ base,
                                            int ld, int m0, int k0, int lane,
                                            float scale) {
  int m = lane & 15, kh = lane >> 4;
  const float* p = base + (size_t)(m0 + m) * ld + k0;
  v16bf a;
#pragma unroll
  for (int j = 0; j < 8; ++j) a[j] = (bf16_t)(p[kh * 8 + j] * scale);
#pragma unroll
  for (int j = 0; j < 8; ++j) a[8 + j] = (bf16_t)(p[16 + kh * 8 + j] * scale);
  return a;
}

// B fragment (32x16, K x N) from an N-major ("B-transposed") bf16 matrix:
// bt[n][k].  lane l: n = l&15, elems 0..7 -> K = (l>>4)*8 + j, elems 8..15 -> K = 16 + (l>>4)*8 + j.
__device__ __forceinline__ v16bf load_b_nmajor(const bf16_t* __restrict__ bt,
                                               int ld, int n0, int k0, int lane) {
  int n = lane & 15, kh = lane >> 4;
  const bf16_t* p = bt + (size_t)(n0 + n) * ld + k0;
  v16bf b;
#pragma unroll
  for (int j = 0; j < 8; ++j) { b[j] = p[kh * 8 + j]; b[8 + j] = p[16 + kh * 8 + j]; }
  return b;
}

// ---------------------------------------------------------------------------
// Kernel 1: 1024x1024 fp32 -> N-major bf16 transpose (for WMMA B operands)
// ---------------------------------------------------------------------------
__global__ __launch_bounds__(256) void transpose_cvt_kernel(
    const float* __restrict__ src, bf16_t* __restrict__ dst) {
  __shared__ float t[32][33];
  int bx = blockIdx.x * 32, by = blockIdx.y * 32;
  int tx = threadIdx.x, ty = threadIdx.y;   // block (32,8)
#pragma unroll
  for (int i = 0; i < 32; i += 8)
    t[ty + i][tx] = src[(size_t)(by + ty + i) * DD + bx + tx];
  __syncthreads();
#pragma unroll
  for (int i = 0; i < 32; i += 8)
    dst[(size_t)(bx + ty + i) * DD + by + tx] = (bf16_t)t[tx][ty + i];
}

// ---------------------------------------------------------------------------
// Kernel 2: projection GEMM  Out = X[8192,1024] * W  (bf16 WMMA, f32 acc)
// wave tile 16(M) x 128(N): one A fragment amortized over 8 WMMAs.
// mode 0: Qh[b][h][s][dk] (scale folded = 1/8)   mode 1: Kh same layout
// mode 2: Vt[b][h][dv][s] (store transposed for PV B-operand)
// ---------------------------------------------------------------------------
__global__ __launch_bounds__(256) void proj_gemm_kernel(
    const float* __restrict__ X, const bf16_t* __restrict__ Wt,
    bf16_t* __restrict__ Out, int mode, float scale) {
  int lane = threadIdx.x & 31;
  int w    = threadIdx.x >> 5;                    // 8 waves
  int nbase = blockIdx.x * 128;
  int mbase = blockIdx.y * 128 + w * 16;

  v8f acc[8] = {};
  for (int k0 = 0; k0 < DD; k0 += 32) {
    __builtin_prefetch(X + (size_t)mbase * DD + k0 + 64, 0, 0);
    v16bf a = load_a_f32(X, DD, mbase, k0, lane, scale);
#pragma unroll
    for (int j = 0; j < 8; ++j) {
      v16bf b = load_b_nmajor(Wt, DD, nbase + j * 16, k0, lane);
      acc[j] = wmma_bf16(a, b, acc[j]);
    }
  }

  int half = lane >> 4, nn = lane & 15;
#pragma unroll
  for (int j = 0; j < 8; ++j) {
#pragma unroll
    for (int r = 0; r < 8; ++r) {
      int m = mbase + r + 8 * half;
      int n = nbase + j * 16 + nn;
      int b = m >> 11, s = m & (SS - 1);
      int h = n >> 6,  d = n & 63;
      size_t idx;
      if (mode == 2)  idx = (((size_t)(b * HH + h)) * DV + d) * SS + s;   // Vt
      else            idx = (((size_t)(b * HH + h)) * SS + s) * DK + d;   // Qh/Kh
      Out[idx] = (bf16_t)acc[j][r];
    }
  }
}

// ---------------------------------------------------------------------------
// Kernel 3: flash attention per (b,h).  4 waves/block, wave = 16 query rows,
// 64-key tiles (4 score fragments) to amortize the softmax butterflies.
// Gaussian bias is a constant logit shift -> cancels in softmax (dropped).
// ---------------------------------------------------------------------------
#define FA_WAVES 4
__global__ __launch_bounds__(128) void flash_attn_kernel(
    const bf16_t* __restrict__ Qh, const bf16_t* __restrict__ Kh,
    const bf16_t* __restrict__ Vt, float* __restrict__ Out) {
  __shared__ float Pst[FA_WAVES][16][65];   // 16 rows x 64 keys (+pad)

  int lane = threadIdx.x & 31;
  int w    = threadIdx.x >> 5;
  int idx  = blockIdx.x;            // B*H*(S/64) = 2048 blocks
  int qblk = idx & 31;
  int bh   = idx >> 5;              // 0..63
  int b = bh >> 4, h = bh & 15;
  int qbase = qblk * 64 + w * 16;

  const bf16_t* qp = Qh + (size_t)bh * SS * DK;
  const bf16_t* kp = Kh + (size_t)bh * SS * DK;
  const bf16_t* vp = Vt + (size_t)bh * DV * SS;

  v16bf q0 = load_a_bf16(qp, DK, qbase, 0, lane);
  v16bf q1 = load_a_bf16(qp, DK, qbase, 32, lane);

  v8f acc[4] = {};
  float rmax[8], rsum[8];
#pragma unroll
  for (int r = 0; r < 8; ++r) { rmax[r] = -3.0e38f; rsum[r] = 0.f; }

  int half = lane >> 4, nn = lane & 15;

  for (int kt = 0; kt < SS; kt += 64) {
    __builtin_prefetch(kp + (size_t)(kt + 64) * DK, 0, 0);
    __builtin_prefetch(vp + (size_t)kt + 64, 0, 0);

    // scores: four 16x16 tiles (keys kt+16f .. kt+16f+15), K-dim = DK = 64
    v8f c[4] = {};
#pragma unroll
    for (int f = 0; f < 4; ++f) {
      c[f] = wmma_bf16(q0, load_b_nmajor(kp, DK, kt + f * 16, 0,  lane), c[f]);
      c[f] = wmma_bf16(q1, load_b_nmajor(kp, DK, kt + f * 16, 32, lane), c[f]);
    }

    // online softmax: C-layout -> row m = r + 8*(lane>>4) lives in one
    // 16-lane half; butterfly-xor over {1,2,4,8} reduces per row exactly.
    float p[4][8], alpha[8];
#pragma unroll
    for (int r = 0; r < 8; ++r) {
      float t = fmaxf(fmaxf(c[0][r], c[1][r]), fmaxf(c[2][r], c[3][r]));
#pragma unroll
      for (int d = 1; d < 16; d <<= 1) t = fmaxf(t, __shfl_xor(t, d, 32));
      float nm = fmaxf(rmax[r], t);
      alpha[r] = __expf(rmax[r] - nm);
      rmax[r]  = nm;
      float ps = 0.f;
#pragma unroll
      for (int f = 0; f < 4; ++f) { p[f][r] = __expf(c[f][r] - nm); ps += p[f][r]; }
#pragma unroll
      for (int d = 1; d < 16; d <<= 1) ps += __shfl_xor(ps, d, 32);
      rsum[r] = rsum[r] * alpha[r] + ps;
    }
#pragma unroll
    for (int j = 0; j < 4; ++j)
#pragma unroll
      for (int r = 0; r < 8; ++r) acc[j][r] *= alpha[r];

    // restage P (16 rows x 64 keys) through LDS into the WMMA A layout
#pragma unroll
    for (int r = 0; r < 8; ++r) {
      int row = r + 8 * half;
#pragma unroll
      for (int f = 0; f < 4; ++f) Pst[w][row][f * 16 + nn] = p[f][r];
    }
    __syncthreads();
    v16bf pa0, pa1;
#pragma unroll
    for (int j = 0; j < 8; ++j) {
      pa0[j]     = (bf16_t)Pst[w][nn][half * 8 + j];
      pa0[8 + j] = (bf16_t)Pst[w][nn][16 + half * 8 + j];
      pa1[j]     = (bf16_t)Pst[w][nn][32 + half * 8 + j];
      pa1[8 + j] = (bf16_t)Pst[w][nn][48 + half * 8 + j];
    }
    __syncthreads();

    // out += P(16x64) * V(64x64): Vt is dv-major so B loads are contiguous
#pragma unroll
    for (int j = 0; j < 4; ++j) {
      acc[j] = wmma_bf16(pa0, load_b_nmajor(vp, SS, j * 16, kt,      lane), acc[j]);
      acc[j] = wmma_bf16(pa1, load_b_nmajor(vp, SS, j * 16, kt + 32, lane), acc[j]);
    }
  }

  float inv[8];
#pragma unroll
  for (int r = 0; r < 8; ++r) inv[r] = 1.f / rsum[r];
#pragma unroll
  for (int j = 0; j < 4; ++j) {
#pragma unroll
    for (int r = 0; r < 8; ++r) {
      int s   = qbase + r + 8 * half;
      int col = h * DV + j * 16 + nn;
      Out[((size_t)b * SS + s) * DD + col] = acc[j][r] * inv[r];
    }
  }
}

// ---------------------------------------------------------------------------
// Kernel 4: FC GEMM + residual.  fc = attn[8192,1024] * Wfc + q
// wave tile 16 x 128 like the projection GEMM.
// ---------------------------------------------------------------------------
__global__ __launch_bounds__(256) void fc_gemm_kernel(
    const float* __restrict__ A, const bf16_t* __restrict__ Wt,
    const float* __restrict__ Res, float* __restrict__ Out) {
  int lane = threadIdx.x & 31;
  int w    = threadIdx.x >> 5;
  int nbase = blockIdx.x * 128;
  int mbase = blockIdx.y * 128 + w * 16;

  v8f acc[8] = {};
  for (int k0 = 0; k0 < DD; k0 += 32) {
    __builtin_prefetch(A + (size_t)mbase * DD + k0 + 64, 0, 0);
    v16bf a = load_a_f32(A, DD, mbase, k0, lane, 1.0f);
#pragma unroll
    for (int j = 0; j < 8; ++j) {
      v16bf b = load_b_nmajor(Wt, DD, nbase + j * 16, k0, lane);
      acc[j] = wmma_bf16(a, b, acc[j]);
    }
  }

  int half = lane >> 4, nn = lane & 15;
#pragma unroll
  for (int j = 0; j < 8; ++j) {
#pragma unroll
    for (int r = 0; r < 8; ++r) {
      size_t m = mbase + r + 8 * half;
      size_t n = nbase + j * 16 + nn;
      Out[m * DD + n] = acc[j][r] + Res[m * DD + n];
    }
  }
}

// ---------------------------------------------------------------------------
// Kernel 5: LayerNorm over last dim (1024), one block per row
// ---------------------------------------------------------------------------
__global__ __launch_bounds__(256) void layernorm_kernel(
    const float* __restrict__ X, const float* __restrict__ gamma,
    const float* __restrict__ beta, float* __restrict__ Out) {
  __shared__ float red[2][8];
  size_t row = blockIdx.x;
  const float* x = X + row * DD;
  float s = 0.f, ss = 0.f;
#pragma unroll
  for (int i = 0; i < 4; ++i) {
    float v = x[threadIdx.x + i * 256];
    s += v; ss += v * v;
  }
#pragma unroll
  for (int d = 1; d < 32; d <<= 1) { s += __shfl_xor(s, d, 32); ss += __shfl_xor(ss, d, 32); }
  int w = threadIdx.x >> 5, lane = threadIdx.x & 31;
  if (lane == 0) { red[0][w] = s; red[1][w] = ss; }
  __syncthreads();
  if (w == 0) {
    s  = (lane < 8) ? red[0][lane] : 0.f;
    ss = (lane < 8) ? red[1][lane] : 0.f;
#pragma unroll
    for (int d = 1; d < 8; d <<= 1) { s += __shfl_xor(s, d, 32); ss += __shfl_xor(ss, d, 32); }
    if (lane == 0) { red[0][0] = s; red[1][0] = ss; }
  }
  __syncthreads();
  float mu  = red[0][0] * (1.f / DD);
  float var = red[1][0] * (1.f / DD) - mu * mu;
  float rs  = rsqrtf(var + 1e-6f);
#pragma unroll
  for (int i = 0; i < 4; ++i) {
    int c = threadIdx.x + i * 256;
    Out[row * DD + c] = (x[c] - mu) * rs * gamma[c] + beta[c];
  }
}

// ---------------------------------------------------------------------------
// Host launcher
// ---------------------------------------------------------------------------
extern "C" void kernel_launch(void* const* d_in, const int* in_sizes, int n_in,
                              void* d_out, int out_size, void* d_ws, size_t ws_size,
                              hipStream_t stream) {
  const float* q     = (const float*)d_in[0];
  const float* k     = (const float*)d_in[1];
  const float* v     = (const float*)d_in[2];
  const float* Wq    = (const float*)d_in[3];
  const float* Wk    = (const float*)d_in[4];
  const float* Wv    = (const float*)d_in[5];
  const float* Wfc   = (const float*)d_in[6];
  const float* gamma = (const float*)d_in[7];
  const float* beta  = (const float*)d_in[8];

  char* ws = (char*)d_ws;
  bf16_t* WqT  = (bf16_t*)(ws + WQT_OFF);
  bf16_t* WkT  = (bf16_t*)(ws + WKT_OFF);
  bf16_t* WvT  = (bf16_t*)(ws + WVT_OFF);
  bf16_t* WfcT = (bf16_t*)(ws + WFCT_OFF);
  bf16_t* Qh   = (bf16_t*)(ws + QH_OFF);
  bf16_t* Kh   = (bf16_t*)(ws + KH_OFF);
  bf16_t* Vt   = (bf16_t*)(ws + VT_OFF);
  float*  Attn = (float*)(ws + ATTN_OFF);
  float*  Fc   = (float*)(ws + FC_OFF);

  dim3 tb(32, 8), tg(32, 32);
  transpose_cvt_kernel<<<tg, tb, 0, stream>>>(Wq,  WqT);
  transpose_cvt_kernel<<<tg, tb, 0, stream>>>(Wk,  WkT);
  transpose_cvt_kernel<<<tg, tb, 0, stream>>>(Wv,  WvT);
  transpose_cvt_kernel<<<tg, tb, 0, stream>>>(Wfc, WfcT);

  dim3 gg(DD / 128, MROWS / 128);  // (8, 64)
  proj_gemm_kernel<<<gg, 256, 0, stream>>>(q, WqT, Qh, 0, 0.125f);  // 1/sqrt(64)
  proj_gemm_kernel<<<gg, 256, 0, stream>>>(k, WkT, Kh, 1, 1.0f);
  proj_gemm_kernel<<<gg, 256, 0, stream>>>(v, WvT, Vt, 2, 1.0f);

  flash_attn_kernel<<<BB * HH * (SS / 64), 128, 0, stream>>>(Qh, Kh, Vt, Attn);

  fc_gemm_kernel<<<gg, 256, 0, stream>>>(Attn, WfcT, q, Fc);

  layernorm_kernel<<<MROWS, 256, 0, stream>>>(Fc, gamma, beta, (float*)d_out);
}
